// CropMappingSwinTransformer_30116310680011
// MI455X (gfx1250) — matrix operand order repaired and checked by reading, not compiled, not run on hardware
//
#include <hip/hip_runtime.h>

typedef __attribute__((ext_vector_type(16))) __bf16 v16bf;
typedef __attribute__((ext_vector_type(8)))  float  v8f;
typedef __attribute__((ext_vector_type(4)))  int    v4i;

#define MTOK   200704                 // B * H * W tokens (= B*nW*N window tokens)
#define SCALEQ 0.17677669529663687f   // 32^-0.5

// ------------------------------------------------------------- async copy path
#if defined(__has_builtin)
#  if __has_builtin(__builtin_amdgcn_global_load_async_to_lds_b128)
#    define HAVE_ASYNC 1
#  endif
#endif
#ifndef HAVE_ASYNC
#  define HAVE_ASYNC 0
#endif

#if HAVE_ASYNC
typedef __attribute__((address_space(1))) v4i* gv4ip;   // global v4i*
typedef __attribute__((address_space(3))) v4i* lv4ip;   // LDS v4i*
__device__ __forceinline__ void async_cp16(const __bf16* g, __bf16* l) {
  __builtin_amdgcn_global_load_async_to_lds_b128(
      (gv4ip)(void*)const_cast<__bf16*>(g), (lv4ip)(void*)l, 0, 0);
}
__device__ __forceinline__ void wait_async() {
#if defined(__has_builtin) && __has_builtin(__builtin_amdgcn_s_wait_asynccnt)
  __builtin_amdgcn_s_wait_asynccnt(0);
#else
  asm volatile("s_wait_asynccnt 0x0" ::: "memory");
#endif
}
#endif

__device__ __forceinline__ int reg56(int y) { return y < 49 ? 0 : (y < 53 ? 1 : 2); }

// --------------------------------------------- f32 -> bf16 with transpose
// src: K x Nc (row-major).  dst: Nc x K (row-major) == W^T, bf16.
__global__ __launch_bounds__(256)
void cvt_t_kernel(const float* __restrict__ src, __bf16* __restrict__ dst,
                  int K, int Nc) {
  int i = blockIdx.x * 256 + threadIdx.x;
  if (i < K * Nc) {
    int n = i / K, k = i - n * K;
    dst[i] = (__bf16)src[(size_t)k * Nc + n];
  }
}

// ------------------- LayerNorm (+ optional shifted-window partition gather)
__global__ __launch_bounds__(256)
void ln_kernel(const float* __restrict__ x, const float* __restrict__ g,
               const float* __restrict__ bv, __bf16* __restrict__ out,
               int windowed, int shift) {
  int t = blockIdx.x * 8 + (threadIdx.x >> 5);
  int lane = threadIdx.x & 31;
  size_t src;
  if (windowed) {
    int bw = t / 49, nn = t - bw * 49;
    int b = bw >> 6, w64 = bw & 63, wi = w64 >> 3, wj = w64 & 7;
    int pi = nn / 7, pj = nn - pi * 7;
    int y  = (wi * 7 + pi + shift) % 56;
    int xw = (wj * 7 + pj + shift) % 56;
    src = (size_t)b * 3136 + y * 56 + xw;
  } else {
    src = (size_t)t;
  }
  const float* row = x + src * 192;
  float v[6]; float s = 0.f;
#pragma unroll
  for (int j = 0; j < 6; ++j) { v[j] = row[lane + 32 * j]; s += v[j]; }
#pragma unroll
  for (int o = 16; o; o >>= 1) s += __shfl_xor(s, o, 32);
  float mean = s * (1.f / 192.f);
  float vs = 0.f;
#pragma unroll
  for (int j = 0; j < 6; ++j) { float d = v[j] - mean; vs += d * d; }
#pragma unroll
  for (int o = 16; o; o >>= 1) vs += __shfl_xor(vs, o, 32);
  float rstd = rsqrtf(vs * (1.f / 192.f) + 1e-5f);
  __bf16* orow = out + (size_t)t * 192;
#pragma unroll
  for (int j = 0; j < 6; ++j) {
    int c = lane + 32 * j;
    orow[c] = (__bf16)((v[j] - mean) * rstd * g[c] + bv[c]);
  }
}

// -------------------------------------- 128x64-tile bf16 WMMA GEMM + epilogues
// A: M x K (bf16, row-major).  BT: Nc x K (bf16, row-major)  == W^T.
// EPI 0: +bias, scale q, scatter to qkv [bw][s][h][49][32]   (obf)
// EPI 1: +bias, window-reverse/unshift, residual add into x  (of32)
// EPI 2: +bias, exact GELU, bf16 row-major store             (obf)
// EPI 3: +bias, residual add into x token-major              (of32)
template<int EPI>
__global__ __launch_bounds__(256)
void gemm_bf16_k(const __bf16* __restrict__ A, const __bf16* __restrict__ BT,
                 const float* __restrict__ bias, int K, int Nc,
                 __bf16* __restrict__ obf, float* __restrict__ of32, int shift) {
  __shared__ alignas(16) __bf16 As[128 * 32];   // [row][k]
  __shared__ alignas(16) __bf16 Bs[64 * 32];    // [col][k]  (transposed tile)
  const int tid = threadIdx.x, wave = tid >> 5, lane = tid & 31;
  const int bm = blockIdx.y * 128, bn = blockIdx.x * 64;
  const int wr = wave >> 1, wc = wave & 1;      // 4 x 2 wave grid
  const int lo = lane & 15, hi = lane >> 4;
  v8f acc00 = {}, acc01 = {}, acc10 = {}, acc11 = {};
  for (int k0 = 0; k0 < K; k0 += 32) {
#if HAVE_ASYNC
#pragma unroll
    for (int idx = tid; idx < 768; idx += 256) {     // 512 A + 256 B b128 copies
      if (idx < 512) {
        int row = idx >> 2, seg = (idx & 3) * 8;
        async_cp16(A + (size_t)(bm + row) * K + k0 + seg, As + row * 32 + seg);
      } else {
        int j = idx - 512; int row = j >> 2, seg = (j & 3) * 8;
        async_cp16(BT + (size_t)(bn + row) * K + k0 + seg, Bs + row * 32 + seg);
      }
    }
    wait_async();
#else
    {
      const unsigned* Ag = (const unsigned*)(A + (size_t)bm * K + k0);
      const unsigned* Bg = (const unsigned*)(BT + (size_t)bn * K + k0);
      unsigned* AsU = (unsigned*)As; unsigned* BsU = (unsigned*)Bs;
      int sK = K >> 1;
#pragma unroll
      for (int i = tid; i < 2048; i += 256)
        AsU[i] = Ag[(size_t)(i >> 4) * sK + (i & 15)];
#pragma unroll
      for (int i = tid; i < 1024; i += 256)
        BsU[i] = Bg[(size_t)(i >> 4) * sK + (i & 15)];
    }
#endif
    __syncthreads();
    v16bf a0, a1, b0, b1;
    const int r0 = wr * 32 + lo, r1 = r0 + 16;
    const int c0 = wc * 32 + lo, c1 = c0 + 16;
#pragma unroll
    for (int e = 0; e < 16; ++e) {
      int ka = (e & 7) + ((e >> 3) << 4) + (hi << 3);   // A k-map per ISA
      a0[e] = As[r0 * 32 + ka];
      a1[e] = As[r1 * 32 + ka];
      int kb = (e & 7) + ((e >> 3) << 4) + (hi << 3);   // B k-map per ISA
      b0[e] = Bs[c0 * 32 + kb];
      b1[e] = Bs[c1 * 32 + kb];
    }
    acc00 = __builtin_amdgcn_wmma_f32_16x16x32_bf16(false, a0, false, b0, (short)0, acc00, false, false);
    acc01 = __builtin_amdgcn_wmma_f32_16x16x32_bf16(false, a0, false, b1, (short)0, acc01, false, false);
    acc10 = __builtin_amdgcn_wmma_f32_16x16x32_bf16(false, a1, false, b0, (short)0, acc10, false, false);
    acc11 = __builtin_amdgcn_wmma_f32_16x16x32_bf16(false, a1, false, b1, (short)0, acc11, false, false);
    __syncthreads();
  }
  v8f accs[4] = {acc00, acc01, acc10, acc11};
#pragma unroll
  for (int t = 0; t < 4; ++t) {
    int am = t >> 1, bnn = t & 1;
    int ng = bn + wc * 32 + bnn * 16 + lo;
    float bval = bias[ng];
#pragma unroll
    for (int r = 0; r < 8; ++r) {
      int mg = bm + wr * 32 + am * 16 + r + (hi << 3);
      float val = accs[t][r] + bval;
      if constexpr (EPI == 0) {
        if (ng < 192) val *= SCALEQ;                     // scale q after bias
        int s = ng / 192, rem = ng - s * 192, hh = rem >> 5, d = rem & 31;
        int bw = mg / 49, nn = mg - bw * 49;
        obf[((((size_t)bw * 3 + s) * 6 + hh) * 49 + nn) * 32 + d] = (__bf16)val;
      } else if constexpr (EPI == 1) {
        int bw = mg / 49, nn = mg - bw * 49;
        int b = bw >> 6, w64 = bw & 63, wi = w64 >> 3, wj = w64 & 7;
        int pi = nn / 7, pj = nn - pi * 7;
        int y  = (wi * 7 + pi + shift) % 56;
        int xw = (wj * 7 + pj + shift) % 56;
        of32[((size_t)b * 3136 + y * 56 + xw) * 192 + ng] += val;  // bijection
      } else if constexpr (EPI == 2) {
        val = 0.5f * val * (1.f + erff(val * 0.70710678118f));
        obf[(size_t)mg * Nc + ng] = (__bf16)val;
      } else {
        of32[(size_t)mg * 192 + ng] += val;
      }
    }
  }
}

// ------------------------------ fused windowed attention, 1 block = (window, head)
__global__ __launch_bounds__(256)
void attn_kernel(const __bf16* __restrict__ qkv, const float* __restrict__ rpb,
                 __bf16* __restrict__ obuf, int shift) {
  __shared__ alignas(16) __bf16 qs[64 * 32], ks[64 * 32], vs[64 * 32];
  __shared__ float  S[64 * 64];
  __shared__ __bf16 P[64 * 64];
  const int bw = blockIdx.x / 6, h = blockIdx.x - bw * 6;
  const int tid = threadIdx.x, wave = tid >> 5, lane = tid & 31;
  const int lo = lane & 15, hi = lane >> 4;
  const __bf16* qb = qkv + (((size_t)bw * 3 + 0) * 6 + h) * 49 * 32;
  const __bf16* kb = qkv + (((size_t)bw * 3 + 1) * 6 + h) * 49 * 32;
  const __bf16* vb = qkv + (((size_t)bw * 3 + 2) * 6 + h) * 49 * 32;
  unsigned* qsU = (unsigned*)qs; unsigned* ksU = (unsigned*)ks; unsigned* vsU = (unsigned*)vs;
#if HAVE_ASYNC
#pragma unroll
  for (int i = tid; i < 720; i += 256) {              // zero rows 49..63 (3 mats)
    int mtx = i / 240, rem = i - mtx * 240;
    unsigned* d = (mtx == 0) ? qsU : (mtx == 1) ? ksU : vsU;
    d[49 * 16 + rem] = 0u;
  }
#pragma unroll
  for (int idx = tid; idx < 588; idx += 256) {        // 49 rows x 4 segs x 3 mats
    int mtx = idx / 196, rem = idx - mtx * 196;
    int row = rem >> 2, seg = (rem & 3) * 8;
    const __bf16* g = ((mtx == 0) ? qb : (mtx == 1) ? kb : vb) + row * 32 + seg;
    __bf16* l = ((mtx == 0) ? qs : (mtx == 1) ? ks : vs) + row * 32 + seg;
    async_cp16(g, l);
  }
  wait_async();
#else
  const unsigned* qg = (const unsigned*)qb;
  const unsigned* kg = (const unsigned*)kb;
  const unsigned* vg = (const unsigned*)vb;
#pragma unroll
  for (int i = tid; i < 1024; i += 256) {
    bool ok = (i >> 4) < 49;
    qsU[i] = ok ? qg[i] : 0u;
    ksU[i] = ok ? kg[i] : 0u;
    vsU[i] = ok ? vg[i] : 0u;
  }
#endif
  __syncthreads();
  {                                                   // S = q k^T + bias + mask
    const int tm = (wave >> 1) * 16, tn = (wave & 1) * 32;
    v8f acc0 = {}; v8f acc1 = {};
    v16bf a, b0, b1;
#pragma unroll
    for (int e = 0; e < 16; ++e) {
      int ka = (e & 7) + ((e >> 3) << 4) + (hi << 3);
      a[e]  = qs[(tm + lo) * 32 + ka];
      int kbm = (e & 7) + ((e >> 3) << 4) + (hi << 3);
      b0[e] = ks[(tn + lo) * 32 + kbm];
      b1[e] = ks[(tn + 16 + lo) * 32 + kbm];
    }
    acc0 = __builtin_amdgcn_wmma_f32_16x16x32_bf16(false, a, false, b0, (short)0, acc0, false, false);
    acc1 = __builtin_amdgcn_wmma_f32_16x16x32_bf16(false, a, false, b1, (short)0, acc1, false, false);
    const int wi = (bw & 63) >> 3, wj = bw & 7;
#pragma unroll
    for (int half = 0; half < 2; ++half) {
      v8f acc = half ? acc1 : acc0;
      int n = tn + 16 * half + lo;                    // key index
#pragma unroll
      for (int r = 0; r < 8; ++r) {
        int m = tm + r + (hi << 3);                   // query index
        float val = -1e30f;
        if (m < 49 && n < 49) {
          int pi = m / 7, pj = m - pi * 7;
          int qi = n / 7, qj = n - qi * 7;
          val = acc[r] + rpb[((pi - qi + 6) * 13 + (pj - qj + 6)) * 6 + h];
          if (shift) {
            int idm = reg56(wi * 7 + pi) * 3 + reg56(wj * 7 + pj);
            int idn = reg56(wi * 7 + qi) * 3 + reg56(wj * 7 + qj);
            if (idm != idn) val -= 100.f;
          }
        }
        S[m * 64 + n] = val;
      }
    }
  }
  __syncthreads();
  if (tid < 64) {                                     // row softmax
    int m = tid;
    if (m < 49) {
      float mx = -1e30f;
      for (int n = 0; n < 49; ++n) mx = fmaxf(mx, S[m * 64 + n]);
      float sum = 0.f;
      for (int n = 0; n < 49; ++n) { float e = __expf(S[m * 64 + n] - mx); S[m * 64 + n] = e; sum += e; }
      float inv = 1.f / sum;
      for (int n = 0; n < 64; ++n) P[m * 64 + n] = (n < 49) ? (__bf16)(S[m * 64 + n] * inv) : (__bf16)0.f;
    } else {
      for (int n = 0; n < 64; ++n) P[m * 64 + n] = (__bf16)0.f;
    }
  }
  __syncthreads();
  {                                                   // O = P @ V
    const int tm = (wave >> 1) * 16, tn = (wave & 1) * 16;
    v8f acc = {};
#pragma unroll
    for (int kk = 0; kk < 64; kk += 32) {
      v16bf a, b;
#pragma unroll
      for (int e = 0; e < 16; ++e) {
        int ka = kk + (e & 7) + ((e >> 3) << 4) + (hi << 3);
        a[e] = P[(tm + lo) * 64 + ka];
        int kbm = kk + (e & 7) + ((e >> 3) << 4) + (hi << 3);
        b[e] = vs[kbm * 32 + tn + lo];
      }
      acc = __builtin_amdgcn_wmma_f32_16x16x32_bf16(false, a, false, b, (short)0, acc, false, false);
    }
#pragma unroll
    for (int r = 0; r < 8; ++r) {
      int m = tm + r + (hi << 3);
      if (m < 49)
        obuf[((size_t)bw * 49 + m) * 192 + h * 32 + tn + lo] = (__bf16)acc[r];
    }
  }
}

// ---------------------------------------------------------------- launcher
extern "C" void kernel_launch(void* const* d_in, const int* in_sizes, int n_in,
                              void* d_out, int out_size, void* d_ws, size_t ws_size,
                              hipStream_t stream) {
  (void)in_sizes; (void)n_in; (void)out_size; (void)ws_size;
  const float* x_in   = (const float*)d_in[0];
  const float* n1g    = (const float*)d_in[1];
  const float* n1b    = (const float*)d_in[2];
  const float* qkv_w  = (const float*)d_in[3];
  const float* qkv_b  = (const float*)d_in[4];
  const float* rpb    = (const float*)d_in[5];
  const float* proj_w = (const float*)d_in[6];
  const float* proj_b = (const float*)d_in[7];
  const float* n2g    = (const float*)d_in[8];
  const float* n2b    = (const float*)d_in[9];
  const float* fc1_w  = (const float*)d_in[10];
  const float* fc1_b  = (const float*)d_in[11];
  const float* fc2_w  = (const float*)d_in[12];
  const float* fc2_b  = (const float*)d_in[13];
  float* x = (float*)d_out;

  // workspace layout (bf16): big (qkv / mlp hidden), window buf, W^T weights
  __bf16* big  = (__bf16*)d_ws;                     // 200704*768
  __bf16* wbuf = big  + (size_t)MTOK * 768;         // 200704*192
  __bf16* wq   = wbuf + (size_t)MTOK * 192;         // 576*192   (qkv_w^T)
  __bf16* wp   = wq   + 576 * 192;                  // 192*192   (proj_w^T)
  __bf16* w1   = wp   + 192 * 192;                  // 768*192   (fc1_w^T)
  __bf16* w2   = w1   + 768 * 192;                  // 192*768   (fc2_w^T)

  (void)hipMemcpyAsync(x, x_in, (size_t)MTOK * 192 * sizeof(float),
                       hipMemcpyDeviceToDevice, stream);

  for (int i = 0; i < 2; ++i) {
    int shift = (i == 0) ? 0 : 3;
    cvt_t_kernel<<<(192*576 + 255)/256, 256, 0, stream>>>(qkv_w  + (size_t)i*192*576, wq, 192, 576);
    cvt_t_kernel<<<(192*192 + 255)/256, 256, 0, stream>>>(proj_w + (size_t)i*192*192, wp, 192, 192);
    cvt_t_kernel<<<(192*768 + 255)/256, 256, 0, stream>>>(fc1_w  + (size_t)i*192*768, w1, 192, 768);
    cvt_t_kernel<<<(768*192 + 255)/256, 256, 0, stream>>>(fc2_w  + (size_t)i*768*192, w2, 768, 192);

    // LN1 + shifted-window partition -> wbuf (window-token major, bf16)
    ln_kernel<<<MTOK/8, 256, 0, stream>>>(x, n1g + i*192, n1b + i*192, wbuf, 1, shift);
    // qkv GEMM -> big in [bw][s][h][49][32] layout (q pre-scaled)
    gemm_bf16_k<0><<<dim3(576/64, MTOK/128), 256, 0, stream>>>(
        wbuf, wq, qkv_b + i*576, 192, 576, big, nullptr, shift);
    // fused windowed attention -> wbuf (merged heads, window-token major)
    attn_kernel<<<4096*6, 256, 0, stream>>>(big, rpb + (size_t)i*169*6, wbuf, shift);
    // proj GEMM + window reverse + residual -> x
    gemm_bf16_k<1><<<dim3(192/64, MTOK/128), 256, 0, stream>>>(
        wbuf, wp, proj_b + i*192, 192, 192, nullptr, x, shift);
    // LN2 -> wbuf (token major)
    ln_kernel<<<MTOK/8, 256, 0, stream>>>(x, n2g + i*192, n2b + i*192, wbuf, 0, 0);
    // fc1 + GELU -> big (200704 x 768 bf16)
    gemm_bf16_k<2><<<dim3(768/64, MTOK/128), 256, 0, stream>>>(
        wbuf, w1, fc1_b + i*768, 192, 768, big, nullptr, 0);
    // fc2 + residual -> x
    gemm_bf16_k<3><<<dim3(192/64, MTOK/128), 256, 0, stream>>>(
        big, w2, fc2_b + i*192, 768, 192, nullptr, x, 0);
  }
}
